// GraphTransformerModel_6279242187130
// MI455X (gfx1250) — matrix operand order repaired and checked by reading, not compiled, not run on hardware
//
#include <hip/hip_runtime.h>
#include <hip/hip_bf16.h>

typedef __attribute__((ext_vector_type(16))) __bf16 v16bf;
typedef __attribute__((ext_vector_type(8)))  __bf16 v8bf;
typedef __attribute__((ext_vector_type(8)))  float  v8f;

#define N_NODES 50000
#define N_EDGES 800000
#define D_IN    384
#define H1      4
#define C1      64
#define D_HID   256   // H1*C1
#define D_OUT   5
#define M_TILES ((N_NODES + 15) / 16)   // 3125 exactly

// ---------------- async-LDS availability (guarded; sync fallback) ----------------
#if defined(__has_builtin)
#  if __has_builtin(__builtin_amdgcn_global_load_async_to_lds_b128) && \
      __has_builtin(__builtin_amdgcn_s_wait_asynccnt)
#    define HAVE_ASYNC_LDS 1
#  endif
#endif
#ifndef HAVE_ASYNC_LDS
#  define HAVE_ASYNC_LDS 0
#endif

// 128-bit payload type + address-space-qualified pointers, per the builtin's
// signature revealed by clang: (v4i AS1*, v4i AS3*, imm offset, imm cpol)
typedef int v4i_ __attribute__((vector_size(16)));
typedef __attribute__((address_space(1))) v4i_* g_v4i_p;
typedef __attribute__((address_space(3))) v4i_* l_v4i_p;

// copy one 16-byte chunk global -> LDS
static __device__ __forceinline__ void copy16_g2l(const __bf16* g, __bf16* l) {
#if HAVE_ASYNC_LDS
    __builtin_amdgcn_global_load_async_to_lds_b128((g_v4i_p)g, (l_v4i_p)l, 0, 0);
#else
    *(v8bf*)l = *(const v8bf*)g;
#endif
}
static __device__ __forceinline__ void wait_async_lds() {
#if HAVE_ASYNC_LDS
    __builtin_amdgcn_s_wait_asynccnt(0);
#endif
}

// ---------- helpers: order-preserving float<->uint for atomicMax ----------
static __device__ __forceinline__ unsigned f2ord(float f) {
    unsigned u = __float_as_uint(f);
    return (u & 0x80000000u) ? ~u : (u | 0x80000000u);
}
static __device__ __forceinline__ float ord2f(unsigned u) {
    unsigned v = (u & 0x80000000u) ? (u & 0x7FFFFFFFu) : ~u;
    return __uint_as_float(v);
}
#define ORD_NEG_INF 0x007FFFFFu   // f2ord(-inf)

// ---------------------------- utility kernels ----------------------------
__global__ void f32_to_bf16_k(const float* __restrict__ in, __bf16* __restrict__ out, int n) {
    int i = blockIdx.x * blockDim.x + threadIdx.x;
    if (i < n) out[i] = (__bf16)in[i];
}

// pack [256 x 5] fp32 weight into [256 x 16] bf16, zero-padded cols 5..15
__global__ void pack_w2_k(const float* __restrict__ W, __bf16* __restrict__ out) {
    int i = blockIdx.x * blockDim.x + threadIdx.x;  // 256*16
    if (i >= D_HID * 16) return;
    int k = i >> 4, n = i & 15;
    out[i] = (n < D_OUT) ? (__bf16)W[k * D_OUT + n] : (__bf16)0.0f;
}

__global__ void fill_u32_k(unsigned* __restrict__ p, unsigned v, int n) {
    int i = blockIdx.x * blockDim.x + threadIdx.x;
    if (i < n) p[i] = v;
}

// ------------------------- layer-1 fused QKVS GEMM ------------------------
// X[50000x384] bf16  x  W[384x256] bf16 (x4 matrices) -> fp32 + bias.
// Block = 256 threads (8 waves = 8 M-tiles) sharing one double-buffered
// 32(k) x 64(n) bf16 weight slab staged in LDS by async DMA.
// grid: (ceil(3125/8), 16).
#define G1_WAVES 8
#define SLAB_LD  72   // padded row stride in elements (144B) to spread DS banks

__global__ __launch_bounds__(256) void gemm1_wmma(
    const __bf16* __restrict__ X,
    const __bf16* __restrict__ W0, const __bf16* __restrict__ W1,
    const __bf16* __restrict__ W2, const __bf16* __restrict__ W3,
    const float* __restrict__ b0, const float* __restrict__ b1,
    const float* __restrict__ b2, const float* __restrict__ b3,
    float* __restrict__ O0, float* __restrict__ O1,
    float* __restrict__ O2, float* __restrict__ O3)
{
    __shared__ __bf16 lds[2][32 * SLAB_LD];

    const int tid   = threadIdx.x;
    const int lane  = tid & 31;
    const int wave  = tid >> 5;
    const int mtile = blockIdx.x * G1_WAVES + wave;
    const bool valid = (mtile < M_TILES);
    const int m0 = (valid ? mtile : (M_TILES - 1)) * 16;

    const int gcol0 = blockIdx.y * 64;       // 0..1023 step 64
    const int mat   = gcol0 >> 8;            // which of the 4 weight matrices
    const int n0    = gcol0 & 255;           // column base within matrix

    const __bf16* B   = (mat == 0) ? W0 : (mat == 1) ? W1 : (mat == 2) ? W2 : W3;
    const float* bias = (mat == 0) ? b0 : (mat == 1) ? b1 : (mat == 2) ? b2 : b3;
    float*       Out  = (mat == 0) ? O0 : (mat == 1) ? O1 : (mat == 2) ? O2 : O3;

    // cooperative DMA mapping: each thread owns one 16B chunk of the slab
    const int cr = tid >> 3;                 // slab row 0..31
    const int cc = (tid & 7) * 8;            // slab col (elements) 0..56
    const __bf16* gsrc = B + (size_t)cr * D_HID + n0 + cc;
    __bf16* ldst[2] = { &lds[0][cr * SLAB_LD + cc], &lds[1][cr * SLAB_LD + cc] };

    // prologue: stage slab for kt=0 into buffer 0
    copy16_g2l(gsrc, ldst[0]);

    // A fragment layout (16-bit A 16x32): lanes 0-15 -> K[0..7],K[16..23];
    // lanes 16-31 -> K[8..15],K[24..31]; row m = lane%16.
    const int lm    = lane & 15;
    const int khalf = (lane < 16) ? 0 : 8;
    const __bf16* arow = X + (size_t)(m0 + lm) * D_IN + khalf;

    v8f acc0 = {}, acc1 = {}, acc2 = {}, acc3 = {};

    for (int kt = 0; kt < D_IN / 32; ++kt) {
        wait_async_lds();          // this wave's chunk of slab(kt) landed
        __syncthreads();           // every wave's chunk landed; prev buffer free

        if (kt + 1 < D_IN / 32)    // overlap next slab DMA with WMMAs
            copy16_g2l(gsrc + (size_t)(kt + 1) * 32 * D_HID, ldst[(kt + 1) & 1]);

        const int k0 = kt * 32;
        v8bf alo = *(const v8bf*)(arow + k0);
        v8bf ahi = *(const v8bf*)(arow + k0 + 16);
        v16bf a;
#pragma unroll
        for (int i = 0; i < 8; ++i) { a[i] = alo[i]; a[i + 8] = ahi[i]; }

        // B fragment (32x16): lane holds col lane%16, slab rows khalf+0..7, 16+khalf+0..7
        const __bf16* lcol = &lds[kt & 1][lm];
        v16bf bb0, bb1, bb2, bb3;
#pragma unroll
        for (int i = 0; i < 8; ++i) {
            const __bf16* r0 = lcol + (khalf + i) * SLAB_LD;
            const __bf16* r1 = lcol + (16 + khalf + i) * SLAB_LD;
            bb0[i] = r0[0];  bb0[i + 8] = r1[0];
            bb1[i] = r0[16]; bb1[i + 8] = r1[16];
            bb2[i] = r0[32]; bb2[i + 8] = r1[32];
            bb3[i] = r0[48]; bb3[i + 8] = r1[48];
        }

        acc0 = __builtin_amdgcn_wmma_f32_16x16x32_bf16(false, a, false, bb0, (short)0, acc0, false, false);
        acc1 = __builtin_amdgcn_wmma_f32_16x16x32_bf16(false, a, false, bb1, (short)0, acc1, false, false);
        acc2 = __builtin_amdgcn_wmma_f32_16x16x32_bf16(false, a, false, bb2, (short)0, acc2, false, false);
        acc3 = __builtin_amdgcn_wmma_f32_16x16x32_bf16(false, a, false, bb3, (short)0, acc3, false, false);
    }

    if (!valid) return;

    // C/D layout: VGPR i, lanes 0-15 -> M=i, N=lane; lanes 16-31 -> M=8+i
    const int nlane = lane & 15;
    const int mbase = m0 + ((lane < 16) ? 0 : 8);
#pragma unroll
    for (int i = 0; i < 8; ++i) {
        float* orow = Out + (size_t)(mbase + i) * D_HID + n0 + nlane;
        orow[ 0] = acc0[i] + bias[n0 +  0 + nlane];
        orow[16] = acc1[i] + bias[n0 + 16 + nlane];
        orow[32] = acc2[i] + bias[n0 + 32 + nlane];
        orow[48] = acc3[i] + bias[n0 + 48 + nlane];
    }
}

// ------------------------- layer-2 fused QKVS GEMM ------------------------
// H[50000x256] bf16 x W2p[4][256x16] bf16 -> 4 outputs [N x 5] fp32 + bias
// Weights are 32KB total (L0-resident): direct global fragment loads.
__global__ __launch_bounds__(128) void gemm2_wmma(
    const __bf16* __restrict__ Hb,
    const __bf16* __restrict__ W2p,     // 4 packed matrices, 256*16 each
    const float* __restrict__ b0, const float* __restrict__ b1,
    const float* __restrict__ b2, const float* __restrict__ b3,
    float* __restrict__ O0, float* __restrict__ O1,
    float* __restrict__ O2, float* __restrict__ O3)
{
    const int lane  = threadIdx.x & 31;
    const int wave  = threadIdx.x >> 5;
    const int mtile = blockIdx.x * 4 + wave;
    if (mtile >= M_TILES) return;
    const int m0 = mtile * 16;

    v8f acc0 = {}, acc1 = {}, acc2 = {}, acc3 = {};
    const int lm    = lane & 15;
    const int khalf = (lane < 16) ? 0 : 8;
    const __bf16* arow = Hb + (size_t)(m0 + lm) * D_HID + khalf;

    for (int k0 = 0; k0 < D_HID; k0 += 32) {
        v8bf alo = *(const v8bf*)(arow + k0);
        v8bf ahi = *(const v8bf*)(arow + k0 + 16);
        v16bf a;
#pragma unroll
        for (int i = 0; i < 8; ++i) { a[i] = alo[i]; a[i + 8] = ahi[i]; }

        const __bf16* bcol = W2p + (size_t)k0 * 16 + lm;
        v16bf bb0, bb1, bb2, bb3;
#pragma unroll
        for (int i = 0; i < 8; ++i) {
            const __bf16* r0 = bcol + (khalf + i) * 16;
            const __bf16* r1 = bcol + (16 + khalf + i) * 16;
            bb0[i] = r0[0 * D_HID * 16]; bb0[i + 8] = r1[0 * D_HID * 16];
            bb1[i] = r0[1 * D_HID * 16]; bb1[i + 8] = r1[1 * D_HID * 16];
            bb2[i] = r0[2 * D_HID * 16]; bb2[i + 8] = r1[2 * D_HID * 16];
            bb3[i] = r0[3 * D_HID * 16]; bb3[i + 8] = r1[3 * D_HID * 16];
        }

        acc0 = __builtin_amdgcn_wmma_f32_16x16x32_bf16(false, a, false, bb0, (short)0, acc0, false, false);
        acc1 = __builtin_amdgcn_wmma_f32_16x16x32_bf16(false, a, false, bb1, (short)0, acc1, false, false);
        acc2 = __builtin_amdgcn_wmma_f32_16x16x32_bf16(false, a, false, bb2, (short)0, acc2, false, false);
        acc3 = __builtin_amdgcn_wmma_f32_16x16x32_bf16(false, a, false, bb3, (short)0, acc3, false, false);
    }

    const int nlane = lane & 15;
    const int mbase = m0 + ((lane < 16) ? 0 : 8);
    if (nlane < D_OUT) {
#pragma unroll
        for (int i = 0; i < 8; ++i) {
            size_t r = (size_t)(mbase + i) * D_OUT + nlane;
            O0[r] = acc0[i] + b0[nlane];
            O1[r] = acc1[i] + b1[nlane];
            O2[r] = acc2[i] + b2[nlane];
            O3[r] = acc3[i] + b3[nlane];
        }
    }
}

// ----------------------------- edge kernels L1 ----------------------------
__global__ void edge_logits1(const float* __restrict__ Q, const float* __restrict__ Kk,
                             const int* __restrict__ src, const int* __restrict__ dst,
                             float* __restrict__ logits, unsigned* __restrict__ mx)
{
    int idx = blockIdx.x * blockDim.x + threadIdx.x;   // e*4 + h
    if (idx >= N_EDGES * H1) return;
    int e = idx >> 2, h = idx & 3;
    int d = dst[e], s = src[e];
    const float4* qp = (const float4*)(Q  + (size_t)d * D_HID + h * C1);
    const float4* kp = (const float4*)(Kk + (size_t)s * D_HID + h * C1);
    float acc = 0.f;
#pragma unroll
    for (int i = 0; i < C1 / 4; ++i) {
        float4 qv = qp[i], kv = kp[i];
        acc += qv.x * kv.x + qv.y * kv.y + qv.z * kv.z + qv.w * kv.w;
    }
    float lg = acc * 0.125f;   // 1/sqrt(64)
    logits[idx] = lg;
    atomicMax(mx + d * H1 + h, f2ord(lg));
}

__global__ void edge_exp1(const int* __restrict__ dst, float* __restrict__ logits /*in/out*/,
                          const unsigned* __restrict__ mx, float* __restrict__ ssum)
{
    int idx = blockIdx.x * blockDim.x + threadIdx.x;
    if (idx >= N_EDGES * H1) return;
    int e = idx >> 2, h = idx & 3;
    int d = dst[e];
    float m = ord2f(mx[d * H1 + h]);
    if (!isfinite(m)) m = 0.f;
    float ex = __expf(logits[idx] - m);
    logits[idx] = ex;
    atomicAdd(ssum + d * H1 + h, ex);
}

__global__ void edge_scatter1(const int* __restrict__ src, const int* __restrict__ dst,
                              const float* __restrict__ V, const float* __restrict__ ebuf,
                              const float* __restrict__ ssum, float* __restrict__ accum)
{
    int idx = blockIdx.x * blockDim.x + threadIdx.x;   // e*256 + c
    if (idx >= N_EDGES * D_HID) return;
    int e = idx >> 8, c = idx & 255, h = c >> 6;
    int d = dst[e], s = src[e];
    float w = ebuf[e * H1 + h] / (ssum[d * H1 + h] + 1e-16f);
    atomicAdd(accum + (size_t)d * D_HID + c, V[(size_t)s * D_HID + c] * w);
}

__global__ void finish1(const float* __restrict__ accum, const float* __restrict__ Sl,
                        __bf16* __restrict__ Hb)
{
    int idx = blockIdx.x * blockDim.x + threadIdx.x;
    if (idx >= N_NODES * D_HID) return;
    float v = accum[idx] + Sl[idx];          // skip already has bias folded in
    v = v > 0.f ? v : 0.f;                   // ReLU
    Hb[idx] = (__bf16)v;
}

// ----------------------------- edge kernels L2 ----------------------------
__global__ void edge_logits2(const float* __restrict__ Q, const float* __restrict__ Kk,
                             const int* __restrict__ src, const int* __restrict__ dst,
                             float* __restrict__ logits, unsigned* __restrict__ mx)
{
    int e = blockIdx.x * blockDim.x + threadIdx.x;
    if (e >= N_EDGES) return;
    int d = dst[e], s = src[e];
    const float* qp = Q + (size_t)d * D_OUT;
    const float* kp = Kk + (size_t)s * D_OUT;
    float acc = 0.f;
#pragma unroll
    for (int i = 0; i < D_OUT; ++i) acc += qp[i] * kp[i];
    float lg = acc * 0.44721359549995793f;   // 1/sqrt(5)
    logits[e] = lg;
    atomicMax(mx + d, f2ord(lg));
}

__global__ void edge_exp2(const int* __restrict__ dst, float* __restrict__ logits,
                          const unsigned* __restrict__ mx, float* __restrict__ ssum)
{
    int e = blockIdx.x * blockDim.x + threadIdx.x;
    if (e >= N_EDGES) return;
    int d = dst[e];
    float m = ord2f(mx[d]);
    if (!isfinite(m)) m = 0.f;
    float ex = __expf(logits[e] - m);
    logits[e] = ex;
    atomicAdd(ssum + d, ex);
}

__global__ void edge_scatter2(const int* __restrict__ src, const int* __restrict__ dst,
                              const float* __restrict__ V, const float* __restrict__ ebuf,
                              const float* __restrict__ ssum, float* __restrict__ accum)
{
    int idx = blockIdx.x * blockDim.x + threadIdx.x;   // e*5 + c
    if (idx >= N_EDGES * D_OUT) return;
    int e = idx / D_OUT, c = idx - e * D_OUT;
    int d = dst[e], s = src[e];
    float w = ebuf[e] / (ssum[d] + 1e-16f);
    atomicAdd(accum + (size_t)d * D_OUT + c, V[(size_t)s * D_OUT + c] * w);
}

__global__ void finish2(const float* __restrict__ accum, const float* __restrict__ Sl,
                        float* __restrict__ out)
{
    int idx = blockIdx.x * blockDim.x + threadIdx.x;
    if (idx >= N_NODES * D_OUT) return;
    out[idx] = accum[idx] + Sl[idx];
}

// --------------------------------- launch ---------------------------------
static inline int cdiv(long long a, int b) { return (int)((a + b - 1) / b); }

extern "C" void kernel_launch(void* const* d_in, const int* in_sizes, int n_in,
                              void* d_out, int out_size, void* d_ws, size_t ws_size,
                              hipStream_t stream) {
    const float* x   = (const float*)d_in[0];
    const int*   ei  = (const int*)d_in[1];
    const int*   src = ei;
    const int*   dst = ei + N_EDGES;
    const float* Wq1 = (const float*)d_in[2];  const float* bq1 = (const float*)d_in[3];
    const float* Wk1 = (const float*)d_in[4];  const float* bk1 = (const float*)d_in[5];
    const float* Wv1 = (const float*)d_in[6];  const float* bv1 = (const float*)d_in[7];
    const float* Ws1 = (const float*)d_in[8];  const float* bs1 = (const float*)d_in[9];
    const float* Wq2 = (const float*)d_in[10]; const float* bq2 = (const float*)d_in[11];
    const float* Wk2 = (const float*)d_in[12]; const float* bk2 = (const float*)d_in[13];
    const float* Wv2 = (const float*)d_in[14]; const float* bv2 = (const float*)d_in[15];
    const float* Ws2 = (const float*)d_in[16]; const float* bs2 = (const float*)d_in[17];

    char* p = (char*)d_ws;
    auto carve = [&](size_t bytes) -> char* {
        char* r = p;
        p += (bytes + 255) & ~(size_t)255;
        return r;
    };

    __bf16* xb   = (__bf16*)carve((size_t)N_NODES * D_IN * 2);
    __bf16* wq1b = (__bf16*)carve((size_t)D_IN * D_HID * 2);
    __bf16* wk1b = (__bf16*)carve((size_t)D_IN * D_HID * 2);
    __bf16* wv1b = (__bf16*)carve((size_t)D_IN * D_HID * 2);
    __bf16* ws1b = (__bf16*)carve((size_t)D_IN * D_HID * 2);
    __bf16* w2p  = (__bf16*)carve((size_t)4 * D_HID * 16 * 2);
    float* Q1  = (float*)carve((size_t)N_NODES * D_HID * 4);
    float* K1  = (float*)carve((size_t)N_NODES * D_HID * 4);
    float* V1  = (float*)carve((size_t)N_NODES * D_HID * 4);
    float* S1  = (float*)carve((size_t)N_NODES * D_HID * 4);
    float* lg1 = (float*)carve((size_t)N_EDGES * H1 * 4);       // logits -> exp in place
    unsigned* mx1 = (unsigned*)carve((size_t)N_NODES * H1 * 4);
    float* ss1 = (float*)carve((size_t)N_NODES * H1 * 4);
    float* ac1 = (float*)carve((size_t)N_NODES * D_HID * 4);
    __bf16* hb = (__bf16*)carve((size_t)N_NODES * D_HID * 2);
    float* Q2  = (float*)carve((size_t)N_NODES * D_OUT * 4);
    float* K2  = (float*)carve((size_t)N_NODES * D_OUT * 4);
    float* V2  = (float*)carve((size_t)N_NODES * D_OUT * 4);
    float* S2  = (float*)carve((size_t)N_NODES * D_OUT * 4);
    float* lg2 = (float*)carve((size_t)N_EDGES * 4);
    unsigned* mx2 = (unsigned*)carve((size_t)N_NODES * 4);
    float* ss2 = (float*)carve((size_t)N_NODES * 4);
    float* ac2 = (float*)carve((size_t)N_NODES * D_OUT * 4);

    // ---- precision conversion / weight packing ----
    f32_to_bf16_k<<<cdiv((long long)N_NODES * D_IN, 256), 256, 0, stream>>>(x, xb, N_NODES * D_IN);
    f32_to_bf16_k<<<cdiv(D_IN * D_HID, 256), 256, 0, stream>>>(Wq1, wq1b, D_IN * D_HID);
    f32_to_bf16_k<<<cdiv(D_IN * D_HID, 256), 256, 0, stream>>>(Wk1, wk1b, D_IN * D_HID);
    f32_to_bf16_k<<<cdiv(D_IN * D_HID, 256), 256, 0, stream>>>(Wv1, wv1b, D_IN * D_HID);
    f32_to_bf16_k<<<cdiv(D_IN * D_HID, 256), 256, 0, stream>>>(Ws1, ws1b, D_IN * D_HID);
    pack_w2_k<<<cdiv(D_HID * 16, 256), 256, 0, stream>>>(Wq2, w2p + 0 * D_HID * 16);
    pack_w2_k<<<cdiv(D_HID * 16, 256), 256, 0, stream>>>(Wk2, w2p + 1 * D_HID * 16);
    pack_w2_k<<<cdiv(D_HID * 16, 256), 256, 0, stream>>>(Wv2, w2p + 2 * D_HID * 16);
    pack_w2_k<<<cdiv(D_HID * 16, 256), 256, 0, stream>>>(Ws2, w2p + 3 * D_HID * 16);

    // ---- per-launch accumulator init (deterministic across graph replays) ----
    (void)hipMemsetAsync(ss1, 0, (size_t)N_NODES * H1 * 4, stream);
    (void)hipMemsetAsync(ac1, 0, (size_t)N_NODES * D_HID * 4, stream);
    (void)hipMemsetAsync(ss2, 0, (size_t)N_NODES * 4, stream);
    (void)hipMemsetAsync(ac2, 0, (size_t)N_NODES * D_OUT * 4, stream);
    fill_u32_k<<<cdiv(N_NODES * H1, 256), 256, 0, stream>>>(mx1, ORD_NEG_INF, N_NODES * H1);
    fill_u32_k<<<cdiv(N_NODES, 256), 256, 0, stream>>>(mx2, ORD_NEG_INF, N_NODES);

    // ---- layer 1 ----
    dim3 g1(cdiv(M_TILES, G1_WAVES), 16);
    gemm1_wmma<<<g1, 256, 0, stream>>>(xb, wq1b, wk1b, wv1b, ws1b,
                                       bq1, bk1, bv1, bs1, Q1, K1, V1, S1);
    edge_logits1<<<cdiv((long long)N_EDGES * H1, 256), 256, 0, stream>>>(Q1, K1, src, dst, lg1, mx1);
    edge_exp1<<<cdiv((long long)N_EDGES * H1, 256), 256, 0, stream>>>(dst, lg1, mx1, ss1);
    edge_scatter1<<<cdiv((long long)N_EDGES * D_HID, 256), 256, 0, stream>>>(src, dst, V1, lg1, ss1, ac1);
    finish1<<<cdiv((long long)N_NODES * D_HID, 256), 256, 0, stream>>>(ac1, S1, hb);

    // ---- layer 2 ----
    gemm2_wmma<<<cdiv(M_TILES, 4), 128, 0, stream>>>(hb, w2p, bq2, bk2, bv2, bs2, Q2, K2, V2, S2);
    edge_logits2<<<cdiv(N_EDGES, 256), 256, 0, stream>>>(Q2, K2, src, dst, lg2, mx2);
    edge_exp2<<<cdiv(N_EDGES, 256), 256, 0, stream>>>(dst, lg2, mx2, ss2);
    edge_scatter2<<<cdiv((long long)N_EDGES * D_OUT, 256), 256, 0, stream>>>(src, dst, V2, lg2, ss2, ac2);
    finish2<<<cdiv((long long)N_NODES * D_OUT, 256), 256, 0, stream>>>(ac2, S2, (float*)d_out);
}
